// Transformer_63101659513116
// MI455X (gfx1250) — matrix operand order repaired and checked by reading, not compiled, not run on hardware
//
#include <hip/hip_runtime.h>
#include <hip/hip_bf16.h>

// ---------------------------------------------------------------------------
// Problem constants (match reference)
// ---------------------------------------------------------------------------
constexpr int Bv = 4;
constexpr int Tv = 1024;
constexpr int Vv = 32000;
constexpr int Cv = 1024;
constexpr int Hv = 16;
constexpr int Lv = 8;
constexpr int HSv = Cv / Hv;      // 64
constexpr int FFv = 4 * Cv;       // 4096
constexpr float EPSv = 1e-5f;
constexpr int MROWS = Bv * Tv;    // 4096
constexpr int KWv = 3 * Cv;       // fused qkv row stride (3072)

// ---------------------------------------------------------------------------
// Types for WMMA
// ---------------------------------------------------------------------------
typedef __attribute__((ext_vector_type(16))) __bf16 bf16x16;
typedef __attribute__((ext_vector_type(8)))  float  v8f;

union FragU {
    unsigned int u[8];
    bf16x16 v;
};

__device__ __forceinline__ unsigned short f2bf(float f) {
    unsigned int u = __float_as_uint(f);
    u += 0x7fffu + ((u >> 16) & 1u);
    return (unsigned short)(u >> 16);
}
__device__ __forceinline__ float bf2f(unsigned short s) {
    return __uint_as_float(((unsigned int)s) << 16);
}

// ---------------------------------------------------------------------------
// fp32 [K][N] -> bf16 transposed [N][K]   (tiled 32x32 via LDS)
// ---------------------------------------------------------------------------
__global__ __launch_bounds__(256) void k_cvt_t(const float* __restrict__ src,
                                               unsigned short* __restrict__ dst,
                                               int K, int N) {
    __shared__ float tile[32][33];
    const int n0 = blockIdx.x * 32;
    const int k0 = blockIdx.y * 32;
    const int tx = threadIdx.x;   // 32
    const int ty = threadIdx.y;   // 8
#pragma unroll
    for (int i = 0; i < 4; ++i)
        tile[ty + i * 8][tx] = src[(size_t)(k0 + ty + i * 8) * N + (n0 + tx)];
    __syncthreads();
#pragma unroll
    for (int i = 0; i < 4; ++i)
        dst[(size_t)(n0 + ty + i * 8) * K + (k0 + tx)] = f2bf(tile[tx][ty + i * 8]);
}

// ---------------------------------------------------------------------------
// Embedding: h[b,t,:] = tok_emb[x[b,t],:] + pos_emb[t,:]
// ---------------------------------------------------------------------------
__global__ void k_embed(const int* __restrict__ x, const float* __restrict__ tok,
                        const float* __restrict__ pos, float* __restrict__ h) {
    const int bt = blockIdx.x;
    const int t = bt % Tv;
    const int tokid = x[bt];
    const float* tr = tok + (size_t)tokid * Cv;
    const float* pr = pos + (size_t)t * Cv;
    float* hr = h + (size_t)bt * Cv;
    for (int c = threadIdx.x; c < Cv; c += blockDim.x)
        hr[c] = tr[c] + pr[c];
}

// ---------------------------------------------------------------------------
// LayerNorm over last dim (C=1024), bf16 output
// ---------------------------------------------------------------------------
__global__ __launch_bounds__(256) void k_ln(const float* __restrict__ h,
                                            const float* __restrict__ w,
                                            const float* __restrict__ b,
                                            unsigned short* __restrict__ out) {
    const int row = blockIdx.x;
    const int tid = threadIdx.x;
    const float* hr = h + (size_t)row * Cv;
    __shared__ float red[256];

    float vals[4];
    float s = 0.f;
#pragma unroll
    for (int i = 0; i < 4; ++i) {
        float v = hr[tid + i * 256];
        vals[i] = v;
        s += v;
    }
    red[tid] = s;
    __syncthreads();
    for (int st = 128; st > 0; st >>= 1) {
        if (tid < st) red[tid] += red[tid + st];
        __syncthreads();
    }
    const float mu = red[0] * (1.0f / Cv);
    __syncthreads();

    float ss = 0.f;
#pragma unroll
    for (int i = 0; i < 4; ++i) {
        float d = vals[i] - mu;
        ss += d * d;
    }
    red[tid] = ss;
    __syncthreads();
    for (int st = 128; st > 0; st >>= 1) {
        if (tid < st) red[tid] += red[tid + st];
        __syncthreads();
    }
    const float rstd = rsqrtf(red[0] * (1.0f / Cv) + EPSv);

    unsigned short* orow = out + (size_t)row * Cv;
#pragma unroll
    for (int i = 0; i < 4; ++i) {
        const int c = tid + i * 256;
        orow[c] = f2bf((vals[i] - mu) * rstd * w[c] + b[c]);
    }
}

// ---------------------------------------------------------------------------
// Tiled bf16 WMMA GEMM:  out[M,N] = A[M,K] @ Wt[N,K]^T   (+bias,+relu,+resid)
// 256 threads = 8 wave32s (4M x 2N). Block tile 256x128, wave tile 64x64
// (4x4 v8f accumulators), BK=64 (two 32-wide WMMA substeps per stage).
// K-major A and Wt -> coalesced uint4 staging, aligned dword LDS stores,
// contiguous dword fragment gathers (4 LDS dwords per WMMA).
// ---------------------------------------------------------------------------
template <bool BIAS, bool RELU, bool RESID, bool OUTF32, bool OUTB16>
__global__ __launch_bounds__(256) void k_gemm(const unsigned short* __restrict__ A,
                                              const unsigned short* __restrict__ Wt,
                                              const float* __restrict__ bias,
                                              float* __restrict__ outF,
                                              unsigned short* __restrict__ outB,
                                              int M, int N, int K) {
    constexpr int BM = 256, BN = 128, BK = 64;
    __shared__ unsigned short As[BM][BK + 2];   // stride 66 ushorts = 33 dwords
    __shared__ unsigned short Bs[BN][BK + 2];

    const int tid  = threadIdx.x;
    const int lane = tid & 31;
    const int wave = tid >> 5;
    const int wM = (wave >> 1) * 64;            // 4 waves along M
    const int wN = (wave & 1) * 64;             // 2 waves along N
    const int ml = lane & 15;
    const int hi = lane >> 4;
    const int m0 = blockIdx.y * BM;
    const int n0 = blockIdx.x * BN;

    v8f acc[4][4] = {};

    for (int k0 = 0; k0 < K; k0 += BK) {
        // ---- stage A tile (256x64) and Wt tile (128x64), both K-major ----
#pragma unroll
        for (int it = 0; it < 8; ++it) {
            const int idx = it * 2048 + tid * 8;
            const int r = idx >> 6;
            const int kk = idx & 63;
            union { uint4 q; unsigned int w[4]; } ua;
            ua.q = *reinterpret_cast<const uint4*>(A + (size_t)(m0 + r) * K + (k0 + kk));
            unsigned int* da = reinterpret_cast<unsigned int*>(&As[r][kk]);
#pragma unroll
            for (int e = 0; e < 4; ++e) da[e] = ua.w[e];
        }
#pragma unroll
        for (int it = 0; it < 4; ++it) {
            const int idx = it * 2048 + tid * 8;
            const int r = idx >> 6;
            const int kk = idx & 63;
            union { uint4 q; unsigned int w[4]; } ub;
            ub.q = *reinterpret_cast<const uint4*>(Wt + (size_t)(n0 + r) * K + (k0 + kk));
            unsigned int* db = reinterpret_cast<unsigned int*>(&Bs[r][kk]);
#pragma unroll
            for (int e = 0; e < 4; ++e) db[e] = ub.w[e];
        }
        if (k0 + BK < K) {  // prefetch next stage (global_prefetch_b8)
            const int idx = tid * 8;
            const int r = idx >> 6;
            const int kk = idx & 63;
            __builtin_prefetch(A  + (size_t)(m0 + r) * K + (k0 + BK + kk), 0, 1);
            __builtin_prefetch(Wt + (size_t)(n0 + r) * K + (k0 + BK + kk), 0, 1);
        }
        __syncthreads();

#pragma unroll
        for (int s = 0; s < 2; ++s) {       // two 32-wide K substeps
            FragU fa[4], fb[4];
#pragma unroll
            for (int i = 0; i < 4; ++i) {
                const unsigned int* ar =
                    reinterpret_cast<const unsigned int*>(&As[wM + i * 16 + ml][0]);
#pragma unroll
                for (int w = 0; w < 4; ++w) {
                    fa[i].u[w]     = ar[s * 16 + hi * 4 + w];
                    fa[i].u[w + 4] = ar[s * 16 + 8 + hi * 4 + w];
                }
            }
#pragma unroll
            for (int j = 0; j < 4; ++j) {
                const unsigned int* br =
                    reinterpret_cast<const unsigned int*>(&Bs[wN + j * 16 + ml][0]);
#pragma unroll
                for (int w = 0; w < 8; ++w) fb[j].u[w] = br[s * 16 + (w >> 2) * 8 + hi * 4 + (w & 3)];
            }
#pragma unroll
            for (int i = 0; i < 4; ++i)
#pragma unroll
                for (int j = 0; j < 4; ++j)
                    acc[i][j] = __builtin_amdgcn_wmma_f32_16x16x32_bf16(
                        false, fa[i].v, false, fb[j].v, (short)0, acc[i][j], false, false);
        }
        __syncthreads();
    }

    // ---- epilogue (C layout: M = r + hi*8, N = ml) ----
#pragma unroll
    for (int i = 0; i < 4; ++i) {
#pragma unroll
        for (int j = 0; j < 4; ++j) {
#pragma unroll
            for (int r = 0; r < 8; ++r) {
                const int row = m0 + wM + i * 16 + hi * 8 + r;
                const int col = n0 + wN + j * 16 + ml;
                float v = acc[i][j][r];
                if (BIAS) v += bias[col];
                if (RELU) v = fmaxf(v, 0.0f);
                const size_t idx = (size_t)row * N + col;
                if (OUTF32) {
                    if (RESID) outF[idx] += v;
                    else       outF[idx]  = v;
                }
                if (OUTB16) outB[idx] = f2bf(v);
            }
        }
    }
}

// ---------------------------------------------------------------------------
// WMMA flash attention. Block = 256 threads = 8 waves; each wave owns a
// 16-query tile of one (b,head); waves share LDS-staged 32-key K/V tiles.
// S = Q@K^T and O += P@V via v_wmma_f32_16x16x32_bf16; online softmax with
// cross-lane (16-lane half) shuffles matching the C-fragment row layout.
// Causal masking applied only on diagonal-straddling tiles (wave-uniform).
// qkv is the fused [M, 3C] bf16 buffer (q | k | v along columns).
// ---------------------------------------------------------------------------
__global__ __launch_bounds__(256) void k_attn(const unsigned short* __restrict__ qkv,
                                              unsigned short* __restrict__ ob) {
    const int h0 = blockIdx.y;
    const int b  = blockIdx.z;
    const int tid  = threadIdx.x;
    const int lane = tid & 31;
    const int wave = tid >> 5;
    const int ml = lane & 15;
    const int hi = lane >> 4;
    const int qb0   = blockIdx.x * 128;     // block's first query
    const int qbase = qb0 + wave * 16;      // wave's first query
    const size_t rowB = (size_t)b * Tv;

    __shared__ unsigned short Ks[32][66];       // [key][d]   (row-major)
    __shared__ unsigned short Vs[64][34];       // [d][key]   (transposed)
    __shared__ unsigned short Ps[8][16][34];    // per-wave P restage [q][key]

    // ---- load Q fragments (16 x 64 over d, A-layout) straight from global ----
    FragU fq[2];
    {
        const unsigned int* q32 = reinterpret_cast<const unsigned int*>(
            qkv + (rowB + qbase + ml) * KWv + (size_t)h0 * HSv);
#pragma unroll
        for (int c = 0; c < 2; ++c)
#pragma unroll
            for (int w = 0; w < 4; ++w) {
                fq[c].u[w]     = q32[c * 16 + hi * 4 + w];
                fq[c].u[w + 4] = q32[c * 16 + 8 + hi * 4 + w];
            }
    }

    v8f accO[4] = {};
    float mrow[8], lrow[8];
#pragma unroll
    for (int r = 0; r < 8; ++r) { mrow[r] = -1e30f; lrow[r] = 0.f; }

    const int jend = qb0 + 128;
    for (int j0 = 0; j0 < jend; j0 += 32) {
        // ---- stage K row-major, V transposed ----
        {
            const int idx = tid * 8;
            const int key = idx >> 6;
            const int dd  = idx & 63;
            const size_t rbase = (rowB + j0 + key) * KWv + (size_t)h0 * HSv + dd;
            union { uint4 q; unsigned int w[4]; } uk;
            union { uint4 q; unsigned short s[8]; } uv;
            uk.q = *reinterpret_cast<const uint4*>(qkv + rbase + Cv);
            uv.q = *reinterpret_cast<const uint4*>(qkv + rbase + 2 * Cv);
            unsigned int* kd = reinterpret_cast<unsigned int*>(&Ks[key][dd]);
#pragma unroll
            for (int e = 0; e < 4; ++e) kd[e] = uk.w[e];
#pragma unroll
            for (int e = 0; e < 8; ++e) Vs[dd + e][key] = uv.s[e];
        }
        __syncthreads();

        if (j0 <= qbase + 15) {   // wave-uniform causal skip (EXEC stays full)
            // ---- S = Q @ K^T (two 16-key subtiles, HS=64 in two k-chunks) ----
            v8f sf[2] = {};
#pragma unroll
            for (int kt = 0; kt < 2; ++kt) {
                const unsigned int* br =
                    reinterpret_cast<const unsigned int*>(&Ks[kt * 16 + ml][0]);
                FragU fk[2];
#pragma unroll
                for (int c = 0; c < 2; ++c)
#pragma unroll
                    for (int w = 0; w < 8; ++w) fk[c].u[w] = br[c * 16 + (w >> 2) * 8 + hi * 4 + (w & 3)];
#pragma unroll
                for (int c = 0; c < 2; ++c)
                    sf[kt] = __builtin_amdgcn_wmma_f32_16x16x32_bf16(
                        false, fq[c].v, false, fk[c].v, (short)0, sf[kt], false, false);
            }
            // ---- scale (+ causal mask only on diagonal tiles) + row-max ----
#pragma unroll
            for (int kt = 0; kt < 2; ++kt)
#pragma unroll
                for (int r = 0; r < 8; ++r) sf[kt][r] *= 0.125f;

            if (j0 + 31 > qbase) {   // wave-uniform: tile straddles the diagonal
#pragma unroll
                for (int kt = 0; kt < 2; ++kt) {
                    const int key = j0 + kt * 16 + ml;
#pragma unroll
                    for (int r = 0; r < 8; ++r) {
                        const int q = qbase + r + hi * 8;
                        if (key > q) sf[kt][r] = -1e30f;
                    }
                }
            }

            float tmax[8];
#pragma unroll
            for (int r = 0; r < 8; ++r) tmax[r] = fmaxf(sf[0][r], sf[1][r]);
#pragma unroll
            for (int r = 0; r < 8; ++r)
#pragma unroll
                for (int xm = 1; xm < 16; xm <<= 1)
                    tmax[r] = fmaxf(tmax[r], __shfl_xor(tmax[r], xm, 32));

            float alpha[8], rsum[8];
#pragma unroll
            for (int r = 0; r < 8; ++r) {
                const float nm = fmaxf(mrow[r], tmax[r]);
                alpha[r] = __expf(mrow[r] - nm);
                mrow[r] = nm;
                rsum[r] = 0.f;
            }
            // ---- P = exp(S - m), restage as bf16 in per-wave LDS ----
#pragma unroll
            for (int kt = 0; kt < 2; ++kt)
#pragma unroll
                for (int r = 0; r < 8; ++r) {
                    const float p = __expf(sf[kt][r] - mrow[r]);
                    rsum[r] += p;
                    Ps[wave][r + hi * 8][kt * 16 + ml] = f2bf(p);
                }
#pragma unroll
            for (int r = 0; r < 8; ++r) {
#pragma unroll
                for (int xm = 1; xm < 16; xm <<= 1)
                    rsum[r] += __shfl_xor(rsum[r], xm, 32);
                lrow[r] = lrow[r] * alpha[r] + rsum[r];
            }
            // ---- rescale O, then O += P @ V ----
#pragma unroll
            for (int dt = 0; dt < 4; ++dt)
#pragma unroll
                for (int r = 0; r < 8; ++r) accO[dt][r] *= alpha[r];

            FragU fp;
            {
                const unsigned int* ar =
                    reinterpret_cast<const unsigned int*>(&Ps[wave][ml][0]);
#pragma unroll
                for (int w = 0; w < 4; ++w) {
                    fp.u[w]     = ar[hi * 4 + w];
                    fp.u[w + 4] = ar[8 + hi * 4 + w];
                }
            }
#pragma unroll
            for (int dt = 0; dt < 4; ++dt) {
                const unsigned int* br =
                    reinterpret_cast<const unsigned int*>(&Vs[dt * 16 + ml][0]);
                FragU fv;
#pragma unroll
                for (int w = 0; w < 8; ++w) fv.u[w] = br[(w >> 2) * 8 + hi * 4 + (w & 3)];
                accO[dt] = __builtin_amdgcn_wmma_f32_16x16x32_bf16(
                    false, fp.v, false, fv.v, (short)0, accO[dt], false, false);
            }
        }
        __syncthreads();
    }

    // ---- write O / l ----
#pragma unroll
    for (int dt = 0; dt < 4; ++dt)
#pragma unroll
        for (int r = 0; r < 8; ++r) {
            const int q = qbase + r + hi * 8;
            const size_t o = (rowB + q) * Cv + (size_t)h0 * HSv + dt * 16 + ml;
            ob[o] = f2bf(accO[dt][r] / lrow[r]);
        }
}

// ---------------------------------------------------------------------------
// Workspace layout (bytes)
// ---------------------------------------------------------------------------
constexpr size_t SZ_H    = (size_t)MROWS * Cv * sizeof(float);   // 16 MiB
constexpr size_t SZ_XB   = (size_t)MROWS * Cv * 2;               // 8 MiB
constexpr size_t SZ_QKV  = (size_t)MROWS * KWv * 2;              // 24 MiB
constexpr size_t SZ_O    = (size_t)MROWS * Cv * 2;               // 8 MiB
constexpr size_t SZ_ACT  = (size_t)MROWS * FFv * 2;              // 32 MiB
constexpr size_t SZ_WQKV = (size_t)KWv * Cv * 2;                 // 6 MiB
constexpr size_t SZ_WO   = (size_t)Cv * Cv * 2;                  // 2 MiB
constexpr size_t SZ_W1   = (size_t)Cv * FFv * 2;                 // 8 MiB
constexpr size_t SZ_W2   = (size_t)Cv * FFv * 2;                 // 8 MiB

constexpr size_t OFF_H    = 0;
constexpr size_t OFF_XB   = OFF_H + SZ_H;
constexpr size_t OFF_QKV  = OFF_XB + SZ_XB;
constexpr size_t OFF_O    = OFF_QKV + SZ_QKV;
constexpr size_t OFF_ACT  = OFF_O + SZ_O;
constexpr size_t OFF_WQKV = OFF_ACT + SZ_ACT;
constexpr size_t OFF_WO   = OFF_WQKV + SZ_WQKV;
constexpr size_t OFF_W1   = OFF_WO + SZ_WO;
constexpr size_t OFF_W2   = OFF_W1 + SZ_W1;
constexpr size_t OFF_WP   = OFF_W2 + SZ_W2;

// ---------------------------------------------------------------------------
extern "C" void kernel_launch(void* const* d_in, const int* in_sizes, int n_in,
                              void* d_out, int out_size, void* d_ws, size_t ws_size,
                              hipStream_t stream) {
    (void)in_sizes; (void)n_in; (void)out_size; (void)ws_size;

    const int*   x       = (const int*)d_in[0];
    const float* tok_emb = (const float*)d_in[1];
    const float* pos_emb = (const float*)d_in[2];
    const float* ln1_w   = (const float*)d_in[3];
    const float* ln1_b   = (const float*)d_in[4];
    const float* wq      = (const float*)d_in[5];
    const float* wk      = (const float*)d_in[6];
    const float* wv      = (const float*)d_in[7];
    const float* wo      = (const float*)d_in[8];
    const float* bo      = (const float*)d_in[9];
    const float* ln2_w   = (const float*)d_in[10];
    const float* ln2_b   = (const float*)d_in[11];
    const float* w1      = (const float*)d_in[12];
    const float* b1      = (const float*)d_in[13];
    const float* w2      = (const float*)d_in[14];
    const float* b2      = (const float*)d_in[15];
    const float* lnf_w   = (const float*)d_in[16];
    const float* lnf_b   = (const float*)d_in[17];
    const float* wp      = (const float*)d_in[18];
    const float* bp      = (const float*)d_in[19];

    char* ws = (char*)d_ws;
    float*          h      = (float*)(ws + OFF_H);
    unsigned short* xb     = (unsigned short*)(ws + OFF_XB);
    unsigned short* qkv_b  = (unsigned short*)(ws + OFF_QKV);
    unsigned short* o_b    = (unsigned short*)(ws + OFF_O);
    unsigned short* act_b  = (unsigned short*)(ws + OFF_ACT);
    unsigned short* wqkv_t = (unsigned short*)(ws + OFF_WQKV);
    unsigned short* wo_t   = (unsigned short*)(ws + OFF_WO);
    unsigned short* w1_t   = (unsigned short*)(ws + OFF_W1);
    unsigned short* w2_t   = (unsigned short*)(ws + OFF_W2);
    unsigned short* wp_t   = (unsigned short*)(ws + OFF_WP);

    const dim3 tcv(32, 8);

    k_embed<<<MROWS, 256, 0, stream>>>(x, tok_emb, pos_emb, h);

    const dim3 gQKV(KWv / 128, MROWS / 256);   // (24, 16)
    const dim3 gCC(Cv / 128, MROWS / 256);     // (8, 16)
    const dim3 gFF(FFv / 128, MROWS / 256);    // (32, 16)
    const dim3 gAtt(Tv / 128, Hv, Bv);         // (8, 16, 4)

    for (int l = 0; l < Lv; ++l) {
        const size_t oCC = (size_t)l * Cv * Cv;
        const size_t oFF = (size_t)l * Cv * FFv;

        // transpose+convert weights: Wt[N][K] bf16 (qkv fused along N rows)
        k_cvt_t<<<dim3(Cv / 32, Cv / 32), tcv, 0, stream>>>(wq + oCC, wqkv_t, Cv, Cv);
        k_cvt_t<<<dim3(Cv / 32, Cv / 32), tcv, 0, stream>>>(wk + oCC, wqkv_t + (size_t)Cv * Cv, Cv, Cv);
        k_cvt_t<<<dim3(Cv / 32, Cv / 32), tcv, 0, stream>>>(wv + oCC, wqkv_t + (size_t)2 * Cv * Cv, Cv, Cv);
        k_cvt_t<<<dim3(Cv / 32, Cv / 32), tcv, 0, stream>>>(wo + oCC, wo_t, Cv, Cv);
        k_cvt_t<<<dim3(FFv / 32, Cv / 32), tcv, 0, stream>>>(w1 + oFF, w1_t, Cv, FFv);
        k_cvt_t<<<dim3(Cv / 32, FFv / 32), tcv, 0, stream>>>(w2 + oFF, w2_t, FFv, Cv);

        // a = LN1(h) -> bf16
        k_ln<<<MROWS, 256, 0, stream>>>(h, ln1_w + (size_t)l * Cv, ln1_b + (size_t)l * Cv, xb);

        // fused qkv = a @ [wq|wk|wv]   (bf16, row stride 3C)
        k_gemm<false, false, false, false, true><<<gQKV, 256, 0, stream>>>(
            xb, wqkv_t, nullptr, nullptr, qkv_b, MROWS, KWv, Cv);

        // WMMA flash attention -> o (bf16)
        k_attn<<<gAtt, 256, 0, stream>>>(qkv_b, o_b);

        // h += o @ wo + bo
        k_gemm<true, false, true, true, false><<<gCC, 256, 0, stream>>>(
            o_b, wo_t, bo + (size_t)l * Cv, h, nullptr, MROWS, Cv, Cv);

        // m = LN2(h) -> bf16
        k_ln<<<MROWS, 256, 0, stream>>>(h, ln2_w + (size_t)l * Cv, ln2_b + (size_t)l * Cv, xb);

        // act = relu(m @ w1 + b1) -> bf16
        k_gemm<true, true, false, false, true><<<gFF, 256, 0, stream>>>(
            xb, w1_t, b1 + (size_t)l * FFv, nullptr, act_b, MROWS, FFv, Cv);

        // h += act @ w2 + b2
        k_gemm<true, false, true, true, false><<<gCC, 256, 0, stream>>>(
            act_b, w2_t, b2 + (size_t)l * Cv, h, nullptr, MROWS, Cv, FFv);
    }

    // final LN -> bf16
    k_ln<<<MROWS, 256, 0, stream>>>(h, lnf_w, lnf_b, xb);

    // LM head: out = hf @ wp + bp  (fp32 into d_out)
    k_cvt_t<<<dim3(Vv / 32, Cv / 32), tcv, 0, stream>>>(wp, wp_t, Cv, Vv);
    const dim3 gLM(Vv / 128, MROWS / 256);     // (250, 16)
    k_gemm<true, false, false, true, false><<<gLM, 256, 0, stream>>>(
        xb, wp_t, bp, (float*)d_out, nullptr, MROWS, Vv, Cv);
}
